// SlidingTileAttention2D_83408264888463
// MI455X (gfx1250) — matrix operand order repaired and checked by reading, not compiled, not run on hardware
//
#include <hip/hip_runtime.h>

// ---------------- CDNA5 WMMA types ----------------
typedef __attribute__((ext_vector_type(16))) __bf16 v16bf;
typedef __attribute__((ext_vector_type(8)))  float  v8f;

union Frag {
    uint4 u[2];      // 2 x 16B = 32B = 16 bf16
    v16bf v;
};

// native bf16 convert (v_cvt on gfx1250), RTNE
__device__ __forceinline__ unsigned short f2bf(float f) {
    __bf16 h = (__bf16)f;
    return __builtin_bit_cast(unsigned short, h);
}

__device__ __forceinline__ float frcp(float f) {
#if defined(__has_builtin)
#if __has_builtin(__builtin_amdgcn_rcpf)
    return __builtin_amdgcn_rcpf(f);
#else
    return 1.f / f;
#endif
#else
    return 1.f / f;
#endif
}

// ---------------- geometry (matches reference) ----------------
#define NB   2
#define SEQ  4096
#define DM   2048
#define NH   32
#define HDIM 64
#define NTW  4
#define NT   32          // tiles
#define TNTOK 128        // tokens per tile
#define QSCALE 0.125f    // 1/sqrt(64), exact power of two

// ---------------- async global->LDS path (gfx1250) ----------------
#if defined(__has_builtin)
#if __has_builtin(__builtin_amdgcn_global_load_async_to_lds_b128) && \
    __has_builtin(__builtin_amdgcn_s_wait_asynccnt)
#define USE_ASYNC 1
#endif
#endif

#if defined(USE_ASYNC)
typedef int v4i_ __attribute__((vector_size(4 * sizeof(int))));
__device__ __forceinline__ void cp16(const unsigned short* g, unsigned short* l) {
    __builtin_amdgcn_global_load_async_to_lds_b128(
        (__attribute__((address_space(1))) v4i_*)(g),
        (__attribute__((address_space(3))) v4i_*)(l), 0, 0);
}
#define WAITA(n) __builtin_amdgcn_s_wait_asynccnt(n)
#else
__device__ __forceinline__ void cp16(const unsigned short* g, unsigned short* l) {
    *(uint4*)l = *(const uint4*)g;
}
#define WAITA(n)
#endif

// ---------------- f32 -> bf16 convert (row-major) ----------------
__global__ void cvt_bf16(const float* __restrict__ in,
                         unsigned short* __restrict__ out, int n8) {
    int i = blockIdx.x * blockDim.x + threadIdx.x;
    if (i >= n8) return;
    const float4* p = (const float4*)in;
    float4 a = p[2 * i], b = p[2 * i + 1];
    uint4 o;
    o.x = (unsigned)f2bf(a.x) | ((unsigned)f2bf(a.y) << 16);
    o.y = (unsigned)f2bf(a.z) | ((unsigned)f2bf(a.w) << 16);
    o.z = (unsigned)f2bf(b.x) | ((unsigned)f2bf(b.y) << 16);
    o.w = (unsigned)f2bf(b.z) | ((unsigned)f2bf(b.w) << 16);
    ((uint4*)out)[i] = o;
}

// ---------------- f32 [K][N] -> bf16 transposed [N][K] (LDS-tiled) --------
__global__ __launch_bounds__(256)
void cvt_wt(const float* __restrict__ W, unsigned short* __restrict__ Wt) {
    __shared__ unsigned short T[64 * 80];          // [n][k], pad 64->80
    const int n0 = blockIdx.x * 64, k0 = blockIdx.y * 64;
    const int tid = threadIdx.x;
#pragma unroll
    for (int c = 0; c < 4; c++) {
        int chunk = c * 256 + tid;                 // 0..1023
        int kr = chunk >> 4, nc = (chunk & 15) * 4;
        float4 d = *(const float4*)(W + (size_t)(k0 + kr) * DM + n0 + nc);
        T[(nc + 0) * 80 + kr] = f2bf(d.x);
        T[(nc + 1) * 80 + kr] = f2bf(d.y);
        T[(nc + 2) * 80 + kr] = f2bf(d.z);
        T[(nc + 3) * 80 + kr] = f2bf(d.w);
    }
    __syncthreads();
#pragma unroll
    for (int c = 0; c < 2; c++) {
        int chunk = c * 256 + tid;                 // 0..511
        int nr = chunk >> 3, kc = (chunk & 7) * 8;
        uint4 d = *(const uint4*)(T + nr * 80 + kc);
        *(uint4*)(Wt + (size_t)(n0 + nr) * DM + k0 + kc) = d;
    }
}

// ---------------- bf16 GEMM: C[8192,2048] = A[8192,2048] * Wt[2048,2048]^T --
// Wt pre-transposed [N][K]. K-depth 64 staging, double-buffered, async.
// MODE 0: bf16 scatter into tile-major [B,H,T,TN,HD]   (Q/K, fused _to_tiles)
// MODE 2: bf16 scatter into transposed  [B,H,T,HD,TN]  (V)
// MODE 1: f32 row-major (final output projection)
template <int MODE>
__global__ __launch_bounds__(256)
void gemm128(const unsigned short* __restrict__ A,
             const unsigned short* __restrict__ Wt,
             unsigned short* __restrict__ Obf,
             float* __restrict__ Of32, float scale) {
    __shared__ unsigned short Al[2][128 * 72];     // [m][k], pad 64->72
    __shared__ unsigned short Bl[2][128 * 72];     // [n][k], pad 64->72

    const int tid  = threadIdx.x;
    const int wave = tid >> 5, lane = tid & 31;
    const int lh = lane >> 4, lm = lane & 15;
    const int wm = wave >> 1, wn = wave & 1;       // 4x2 wave grid, 32x64/wave
    const int m0 = blockIdx.y * 128, n0 = blockIdx.x * 128;

    v8f acc[2][4];
#pragma unroll
    for (int i = 0; i < 2; i++)
#pragma unroll
        for (int j = 0; j < 4; j++) acc[i][j] = {};

    auto stage = [&](int buf, int k0) {
#pragma unroll
        for (int c = 0; c < 4; c++) {
            int idx = c * 256 + tid;               // 0..1023
            int row = idx >> 3, col = (idx & 7) * 8;
            cp16(A + (size_t)(m0 + row) * DM + k0 + col,
                 &Al[buf][row * 72 + col]);
        }
#pragma unroll
        for (int c = 0; c < 4; c++) {
            int idx = c * 256 + tid;
            int nr = idx >> 3, kc = (idx & 7) * 8;
            cp16(Wt + (size_t)(n0 + nr) * DM + k0 + kc,
                 &Bl[buf][nr * 72 + kc]);
        }
    };

    stage(0, 0);
    int p = 0;
    for (int k0 = 0; k0 < DM; k0 += 64) {
        if (k0 + 64 < DM) {
            stage(p ^ 1, k0 + 64);
            WAITA(8);                              // retire buffer p's 8 loads
        } else {
            WAITA(0);
        }
        __syncthreads();

#pragma unroll
        for (int ks = 0; ks < 2; ks++) {
            Frag af[2], bfr[4];
#pragma unroll
            for (int mi = 0; mi < 2; mi++) {
                int row = wm * 32 + mi * 16 + lm;
                af[mi].u[0] = *(const uint4*)(&Al[p][row * 72 + ks * 32 + lh * 8]);
                af[mi].u[1] = *(const uint4*)(&Al[p][row * 72 + ks * 32 + 16 + lh * 8]);
            }
#pragma unroll
            for (int ni = 0; ni < 4; ni++) {
                int nr = wn * 64 + ni * 16 + lm;
                bfr[ni].u[0] = *(const uint4*)(&Bl[p][nr * 72 + ks * 32 + lh * 8]);
                bfr[ni].u[1] = *(const uint4*)(&Bl[p][nr * 72 + ks * 32 + 16 + lh * 8]);
            }
#pragma unroll
            for (int mi = 0; mi < 2; mi++)
#pragma unroll
                for (int ni = 0; ni < 4; ni++)
                    acc[mi][ni] = __builtin_amdgcn_wmma_f32_16x16x32_bf16(
                        false, af[mi].v, false, bfr[ni].v, (short)0,
                        acc[mi][ni], false, false);
        }
        __syncthreads();
        p ^= 1;
    }

    // epilogue
#pragma unroll
    for (int mi = 0; mi < 2; mi++)
#pragma unroll
        for (int ni = 0; ni < 4; ni++)
#pragma unroll
            for (int r = 0; r < 8; r++) {
                int Mrow = m0 + wm * 32 + mi * 16 + lh * 8 + r;
                int Ncol = n0 + wn * 64 + ni * 16 + lm;
                float v = acc[mi][ni][r];
                if (MODE == 1) {
                    Of32[(size_t)Mrow * DM + Ncol] = v;
                } else {
                    v *= scale;
                    int b = Mrow >> 12, pp = Mrow & 4095;
                    int gh = pp >> 6, gw = pp & 63;
                    int t  = (gh >> 3) * NTW + (gw >> 4);
                    int tn = (gh & 7) * 16 + (gw & 15);
                    int h  = Ncol >> 6, hd = Ncol & 63;
                    size_t base = (((size_t)b * NH + h) * NT + t) * (TNTOK * HDIM);
                    size_t dst = (MODE == 0) ? base + (size_t)tn * HDIM + hd
                                             : base + (size_t)hd * TNTOK + tn;
                    Obf[dst] = f2bf(v);
                }
            }
}

// ---------------- block-sparse flash attention --------------------------
// grid = B*H*T; 256 threads = 8 waves; wave w owns query rows [16w,16w+16).
// Qt/Kt tile-major [B,H,T,TN,HD] (Q pre-scaled by 1/8); Vt transposed
// [B,H,T,HD,TN]. KV chunks staged through LDS with async copies.
__global__ __launch_bounds__(256)
void attn(const unsigned short* __restrict__ Qt,
          const unsigned short* __restrict__ Kt,
          const unsigned short* __restrict__ Vt,
          unsigned short* __restrict__ Obf) {
    __shared__ unsigned short Klds[128 * 72];       // [kv][hd] pad 64->72
    __shared__ unsigned short Vlds[64 * 136];       // [hd][kv] pad 128->136
    __shared__ unsigned short Plds[8 * 16 * 40];    // per-wave P staging

    const int blk = blockIdx.x;
    const int t = blk & 31, h = (blk >> 5) & 31, b = blk >> 10;
    const int tid = threadIdx.x, wave = tid >> 5, lane = tid & 31;
    const int lh = lane >> 4, lm = lane & 15;

    const int qh = t >> 2, qw = t & 3;
    const int ch = qh < 1 ? 1 : (qh > 7 ? 7 : qh);
    const int cw = qw < 1 ? 1 : (qw > 3 ? 3 : qw);
    int kts[4];
#pragma unroll
    for (int i = 0; i < 4; i++)
        kts[i] = (ch - 1 + (i >> 1)) * NTW + (cw - 1 + (i & 1));

    const size_t bh = ((size_t)b * NH + h) * NT;
    const unsigned short* qbase = Qt + (bh + t) * (TNTOK * HDIM);

    Frag aq[2];
    {
        const unsigned short* qp = qbase + (wave * 16 + lm) * HDIM;
#pragma unroll
        for (int ks = 0; ks < 2; ks++) {
            aq[ks].u[0] = *(const uint4*)(qp + ks * 32 + lh * 8);
            aq[ks].u[1] = *(const uint4*)(qp + ks * 32 + 16 + lh * 8);
        }
    }

    v8f oacc[4];
#pragma unroll
    for (int i = 0; i < 4; i++) oacc[i] = {};
    float mrun[8], lrun[8];
#pragma unroll
    for (int r = 0; r < 8; r++) { mrun[r] = -1e30f; lrun[r] = 0.f; }

    for (int c = 0; c < 4; c++) {
        const unsigned short* kp = Kt + (bh + kts[c]) * (TNTOK * HDIM);
        const unsigned short* vp = Vt + (bh + kts[c]) * (TNTOK * HDIM);
        __syncthreads();
        // K: 1024 b128 chunks; V (transposed layout): 1024 chunks; 4+4/thread
#pragma unroll
        for (int i = 0; i < 4; i++) {
            int idx = i * 256 + tid;
            int krow = idx >> 3, kc8 = (idx & 7) * 8;
            cp16(kp + krow * HDIM + kc8, &Klds[krow * 72 + kc8]);
            int vrow = idx >> 4, vseg = (idx & 15) * 8;
            cp16(vp + vrow * TNTOK + vseg, &Vlds[vrow * 136 + vseg]);
        }
        WAITA(0);
        __syncthreads();

        // scores S[16 x 128] (Q pre-scaled)
        v8f sc[8];
#pragma unroll
        for (int nt = 0; nt < 8; nt++) {
            v8f s = {};
            int krow = nt * 16 + lm;
#pragma unroll
            for (int ks = 0; ks < 2; ks++) {
                Frag bk;
                bk.u[0] = *(const uint4*)(&Klds[krow * 72 + ks * 32 + lh * 8]);
                bk.u[1] = *(const uint4*)(&Klds[krow * 72 + ks * 32 + 16 + lh * 8]);
                s = __builtin_amdgcn_wmma_f32_16x16x32_bf16(
                    false, aq[ks].v, false, bk.v, (short)0, s, false, false);
            }
            sc[nt] = s;
        }

        // online softmax: row = 8*lh + r, columns spread over 16-lane halves
#pragma unroll
        for (int r = 0; r < 8; r++) {
            float mx = -1e30f;
#pragma unroll
            for (int nt = 0; nt < 8; nt++) mx = fmaxf(mx, sc[nt][r]);
#pragma unroll
            for (int off = 1; off < 16; off <<= 1)
                mx = fmaxf(mx, __shfl_xor(mx, off, 32));
            float mnew  = fmaxf(mrun[r], mx);
            float alpha = __expf(mrun[r] - mnew);
            float rs = 0.f;
#pragma unroll
            for (int nt = 0; nt < 8; nt++) {
                float e = __expf(sc[nt][r] - mnew);
                sc[nt][r] = e;
                rs += e;
            }
#pragma unroll
            for (int off = 1; off < 16; off <<= 1)
                rs += __shfl_xor(rs, off, 32);
            lrun[r] = lrun[r] * alpha + rs;
            mrun[r] = mnew;
#pragma unroll
            for (int ni = 0; ni < 4; ni++) oacc[ni][r] *= alpha;
        }

        // P @ V : 4 kv sub-steps of 32; P goes D-layout -> A-layout via LDS
        unsigned short* pb = Plds + wave * 16 * 40;
#pragma unroll
        for (int kvs = 0; kvs < 4; kvs++) {
#pragma unroll
            for (int nt2 = 0; nt2 < 2; nt2++) {
                int nt = kvs * 2 + nt2;
#pragma unroll
                for (int r = 0; r < 8; r++)
                    pb[(lh * 8 + r) * 40 + nt2 * 16 + lm] = f2bf(sc[nt][r]);
            }
            Frag ap;
            ap.u[0] = *(const uint4*)(pb + lm * 40 + lh * 8);
            ap.u[1] = *(const uint4*)(pb + lm * 40 + 16 + lh * 8);
#pragma unroll
            for (int ni = 0; ni < 4; ni++) {
                Frag bv;
                int vr = ni * 16 + lm;
                bv.u[0] = *(const uint4*)(&Vlds[vr * 136 + kvs * 32 + lh * 8]);
                bv.u[1] = *(const uint4*)(&Vlds[vr * 136 + kvs * 32 + 16 + lh * 8]);
                oacc[ni] = __builtin_amdgcn_wmma_f32_16x16x32_bf16(
                    false, ap.v, false, bv.v, (short)0, oacc[ni], false, false);
            }
        }
    }

    // normalize and write O row-major [B*S, H*HD] bf16 (fused _from_tiles)
    float inv[8];
#pragma unroll
    for (int r = 0; r < 8; r++) inv[r] = frcp(lrun[r]);
#pragma unroll
    for (int ni = 0; ni < 4; ni++)
#pragma unroll
        for (int r = 0; r < 8; r++) {
            int tn = wave * 16 + lh * 8 + r;
            int gh = qh * 8 + (tn >> 4);
            int gw = qw * 16 + (tn & 15);
            int srow = b * SEQ + gh * 64 + gw;
            int col = h * HDIM + ni * 16 + lm;
            Obf[(size_t)srow * DM + col] = f2bf(oacc[ni][r] * inv[r]);
        }
}

// ---------------- launcher ----------------
extern "C" void kernel_launch(void* const* d_in, const int* in_sizes, int n_in,
                              void* d_out, int out_size, void* d_ws, size_t ws_size,
                              hipStream_t stream) {
    const float* X  = (const float*)d_in[0];
    const float* Wq = (const float*)d_in[1];
    const float* Wk = (const float*)d_in[2];
    const float* Wv = (const float*)d_in[3];
    const float* Wo = (const float*)d_in[4];
    float* out = (float*)d_out;

    const size_t NX = (size_t)NB * SEQ * DM;   // 16,777,216 elements
    const size_t NW = (size_t)DM * DM;         //  4,194,304 elements

    unsigned short* ws  = (unsigned short*)d_ws;
    unsigned short* Xbf = ws;
    unsigned short* Wqb = Xbf + NX;            // transposed [N][K]
    unsigned short* Wkb = Wqb + NW;
    unsigned short* Wvb = Wkb + NW;
    unsigned short* Wob = Wvb + NW;
    unsigned short* Qt  = Wob + NW;
    unsigned short* Kt  = Qt + NX;
    unsigned short* Vt  = Kt + NX;
    unsigned short* Ob  = Vt + NX;

    cvt_bf16<<<(int)(NX / 8 / 256), 256, 0, stream>>>(X, Xbf, (int)(NX / 8));
    dim3 wgrid(DM / 64, DM / 64);              // 32 x 32
    cvt_wt<<<wgrid, 256, 0, stream>>>(Wq, Wqb);
    cvt_wt<<<wgrid, 256, 0, stream>>>(Wk, Wkb);
    cvt_wt<<<wgrid, 256, 0, stream>>>(Wv, Wvb);
    cvt_wt<<<wgrid, 256, 0, stream>>>(Wo, Wob);

    dim3 ggrid(DM / 128, (NB * SEQ) / 128);    // 16 x 64
    gemm128<0><<<ggrid, 256, 0, stream>>>(Xbf, Wqb, Qt, nullptr, QSCALE);
    gemm128<0><<<ggrid, 256, 0, stream>>>(Xbf, Wkb, Kt, nullptr, 1.0f);
    gemm128<2><<<ggrid, 256, 0, stream>>>(Xbf, Wvb, Vt, nullptr, 1.0f);

    attn<<<NB * NH * NT, 256, 0, stream>>>(Qt, Kt, Vt, Ob);

    gemm128<1><<<ggrid, 256, 0, stream>>>(Ob, Wob, nullptr, out, 1.0f);
}
